// Net_38809324486772
// MI455X (gfx1250) — compile-verified, matrix operand order checked by reading, not run on hardware
//
#include <hip/hip_runtime.h>

#define N_NODES 100000
#define N_EDGES 3200000

typedef __attribute__((ext_vector_type(16))) _Float16 v16h;
typedef __attribute__((ext_vector_type(8)))  float    v8f;

__device__ __forceinline__ void wave_sync() {
  __builtin_amdgcn_wave_barrier();
  asm volatile("s_wait_dscnt 0x0" ::: "memory");
  __builtin_amdgcn_wave_barrier();
}

__device__ __forceinline__ v8f wmma16(v16h a, v16h b, v8f c) {
  // D(16x16,f32) = A(16x32,f16) * B(32x16,f16) + C   -> v_wmma_f32_16x16x32_f16
  return __builtin_amdgcn_wmma_f32_16x16x32_f16(false, a, false, b, (short)0, c,
                                                false, false);
}

// A-layout K index for element j of the per-lane v16h (ISA 7.12.2, 16-bit A 16x32):
// lanes 0-15 hold K=0..7 (v0..3) and K=16..23 (v4..7); lanes 16-31 hold +8.
__device__ __forceinline__ int kmap(int j, int koff) {
  return (j < 8 ? j : j + 8) + koff;
}

// Apply bias+ReLU to a C/D-layout f32 accumulator, stash as f16 in LDS
// (row-major 16x16), reload in A-layout for the next WMMA (K=16, zero-pad to 32).
__device__ __forceinline__ v16h relu_transpose(_Float16* lh, v8f c, float bias, int lane) {
  const int n = lane & 15, hi = lane >> 4, koff = hi * 8;
  #pragma unroll
  for (int r = 0; r < 8; ++r) {
    float v = c[r] + bias;
    lh[(r + 8 * hi) * 16 + n] = (_Float16)(v > 0.f ? v : 0.f);
  }
  wave_sync();
  const int m = lane & 15;
  v16h a;
  #pragma unroll
  for (int j = 0; j < 16; ++j) {
    int k = kmap(j, koff);
    a[j] = (k < 16) ? lh[m * 16 + k] : (_Float16)0.f;
  }
  return a;
}

// B-layout loader for a (kin x nout) row-major f32 weight, zero-padded to 32x16.
__device__ __forceinline__ v16h load_B(const float* w, int n, int koff, int kbase,
                                       int kin, int nout) {
  v16h b;
  #pragma unroll
  for (int j = 0; j < 16; ++j) {
    int k = kbase + kmap(j, koff);
    b[j] = (k < kin && n < nout) ? (_Float16)w[k * nout + n] : (_Float16)0.f;
  }
  return b;
}

__global__ void zero_kernel(float* __restrict__ p, int n) {
  int i = blockIdx.x * blockDim.x + threadIdx.x;
  if (i < n) p[i] = 0.f;
}

// ---------------- GINE edge stage ----------------
template<int DF>   // per-part feature dim: 3 (layer1) or 16
__global__ void edge_conv(const float* __restrict__ xn,   // N x DF
                          const float* ea,                 // E x DF (may alias eout)
                          const int* __restrict__ src,
                          const int* __restrict__ dst,
                          const float* __restrict__ ew1, const float* __restrict__ eb1,
                          const float* __restrict__ ew2, const float* __restrict__ eb2,
                          float* eout,                     // E x 16
                          float* __restrict__ agg,         // N x DF (atomic)
                          int nTiles) {
  constexpr int KIN = 3 * DF;
  __shared__ float    sfeat[8][16][KIN];
  __shared__ _Float16 shid[8][256];

  const int wave = threadIdx.x >> 5, lane = threadIdx.x & 31;
  const int tile = blockIdx.x * 8 + wave;
  if (tile >= nTiles) return;                 // wave-uniform: EXEC stays all-1s
  const int base = tile * 16;

  // Stage concat features [x_src, e, x_dst] for the 16-edge tile into LDS.
  for (int idx = lane; idx < 16 * KIN; idx += 32) {
    int m = idx / KIN, k = idx % KIN;
    int e = base + m;
    float v;
    if (k < DF)           v = xn[(long)src[e] * DF + k];
    else if (k < 2 * DF)  v = ea[(long)e * DF + (k - DF)];
    else                  v = xn[(long)dst[e] * DF + (k - 2 * DF)];
    sfeat[wave][m][k] = v;
  }
  wave_sync();

  const int m = lane & 15, hi = lane >> 4, koff = hi * 8, n = m;

  // GEMM1: [16 x KIN] @ ew1[KIN x 16]  (KIN=9 -> 1 WMMA, KIN=48 -> 2 WMMAs)
  v8f c = {};
  constexpr int NCH = (KIN + 31) / 32;
  #pragma unroll
  for (int ch = 0; ch < NCH; ++ch) {
    v16h a;
    #pragma unroll
    for (int j = 0; j < 16; ++j) {
      int k = ch * 32 + kmap(j, koff);
      a[j] = (k < KIN) ? (_Float16)sfeat[wave][m][k] : (_Float16)0.f;
    }
    v16h bm = load_B(ew1, n, koff, ch * 32, KIN, 16);
    c = wmma16(a, bm, c);
  }

  // bias+ReLU, LDS transpose, GEMM2: [16x16] @ ew2[16x16], bias+ReLU, store.
  v16h a2  = relu_transpose(shid[wave], c, eb1[n], lane);
  v16h bm2 = load_B(ew2, n, koff, 0, 16, 16);
  v8f c2 = {};
  c2 = wmma16(a2, bm2, c2);

  const float bb = eb2[n];
  #pragma unroll
  for (int r = 0; r < 8; ++r) {
    float v = c2[r] + bb;
    eout[(long)(base + r + 8 * hi) * 16 + n] = (v > 0.f ? v : 0.f);
  }

  // msg = relu(x_src + e) -> atomic scatter-add into agg[dst]
  const int d = dst[base + m];
  if (DF == 16) {
    #pragma unroll
    for (int i = 0; i < 8; ++i) {
      int f = koff + i;
      float v = sfeat[wave][m][f] + sfeat[wave][m][DF + f];
      atomicAdd(&agg[(long)d * DF + f], v > 0.f ? v : 0.f);
    }
  } else if (hi == 0) {
    #pragma unroll
    for (int f = 0; f < DF; ++f) {
      float v = sfeat[wave][m][f] + sfeat[wave][m][DF + f];
      atomicAdd(&agg[(long)d * DF + f], v > 0.f ? v : 0.f);
    }
  }
}

// ---------------- GINE node stage: out = MLP2(relu)(agg + x_prev) ----------------
template<int DIN>  // 3 (layer1) or 16
__global__ void node_conv(const float* xprev,              // N x DIN (may alias xout)
                          const float* __restrict__ agg,   // N x DIN
                          const float* __restrict__ nw1, const float* __restrict__ nb1,
                          const float* __restrict__ nw2, const float* __restrict__ nb2,
                          float* xout,                     // N x 16
                          int nTiles) {
  __shared__ _Float16 shid[8][256];
  const int wave = threadIdx.x >> 5, lane = threadIdx.x & 31;
  const int tile = blockIdx.x * 8 + wave;
  if (tile >= nTiles) return;
  const int base = tile * 16;
  const int m = lane & 15, hi = lane >> 4, koff = hi * 8, n = m;

  v16h a;
  #pragma unroll
  for (int j = 0; j < 16; ++j) {
    int k = kmap(j, koff);
    a[j] = (k < DIN)
         ? (_Float16)(agg[(long)(base + m) * DIN + k] + xprev[(long)(base + m) * DIN + k])
         : (_Float16)0.f;
  }
  v16h bm1 = load_B(nw1, n, koff, 0, DIN, 16);
  v8f c = {};
  c = wmma16(a, bm1, c);

  v16h a2  = relu_transpose(shid[wave], c, nb1[n], lane);
  v16h bm2 = load_B(nw2, n, koff, 0, 16, 16);
  v8f c2 = {};
  c2 = wmma16(a2, bm2, c2);

  const float bb = nb2[n];
  #pragma unroll
  for (int r = 0; r < 8; ++r) {
    float v = c2[r] + bb;
    xout[(long)(base + r + 8 * hi) * 16 + n] = (v > 0.f ? v : 0.f);
  }
}

// ---------------- output heads: Linear(16->16) -> ReLU -> Linear(16->3) ----------------
__global__ void head_kernel(const float* __restrict__ xin,  // R x 16
                            const float* __restrict__ w1, const float* __restrict__ b1,
                            const float* __restrict__ w2, const float* __restrict__ b2,
                            float* __restrict__ out,        // R x 3
                            int nTiles) {
  __shared__ _Float16 shid[8][256];
  const int wave = threadIdx.x >> 5, lane = threadIdx.x & 31;
  const int tile = blockIdx.x * 8 + wave;
  if (tile >= nTiles) return;
  const int base = tile * 16;
  const int m = lane & 15, hi = lane >> 4, koff = hi * 8, n = m;

  v16h a;
  #pragma unroll
  for (int j = 0; j < 16; ++j) {
    int k = kmap(j, koff);
    a[j] = (k < 16) ? (_Float16)xin[(long)(base + m) * 16 + k] : (_Float16)0.f;
  }
  v16h bm1 = load_B(w1, n, koff, 0, 16, 16);
  v8f c = {};
  c = wmma16(a, bm1, c);

  v16h a2  = relu_transpose(shid[wave], c, b1[n], lane);
  v16h bm2 = load_B(w2, n, koff, 0, 16, 3);    // N zero-padded 3 -> 16
  v8f c2 = {};
  c2 = wmma16(a2, bm2, c2);

  const float bb = (n < 3) ? b2[n] : 0.f;
  #pragma unroll
  for (int r = 0; r < 8; ++r) {
    if (n < 3) out[(long)(base + r + 8 * hi) * 3 + n] = c2[r] + bb;  // no final ReLU
  }
}

extern "C" void kernel_launch(void* const* d_in, const int* in_sizes, int n_in,
                              void* d_out, int out_size, void* d_ws, size_t ws_size,
                              hipStream_t stream) {
  (void)in_sizes; (void)n_in; (void)out_size; (void)ws_size;
  auto F = [&](int i) { return (const float*)d_in[i]; };
  const float* x  = F(0);
  const float* ea = F(1);
  const int* eidx = (const int*)d_in[2];
  const int* src = eidx;
  const int* dst = eidx + N_EDGES;

  float* xn  = (float*)d_ws;                       // N x 16
  float* agg = xn + (size_t)N_NODES * 16;          // N x 16
  float* xe  = agg + (size_t)N_NODES * 16;         // E x 16
  float* out = (float*)d_out;

  const int eTiles = N_EDGES / 16, nTiles = N_NODES / 16;
  const int eBlk = (eTiles + 7) / 8, nBlk = (nTiles + 7) / 8;
  const int zBlk = (N_NODES * 16 + 255) / 256;
  dim3 blk(256);

  // ---- layer 1 (DF=3) ----
  zero_kernel<<<zBlk, 256, 0, stream>>>(agg, N_NODES * 16);
  edge_conv<3><<<eBlk, blk, 0, stream>>>(x, ea, src, dst, F(3), F(4), F(5), F(6),
                                         xe, agg, eTiles);
  node_conv<3><<<nBlk, blk, 0, stream>>>(x, agg, F(7), F(8), F(9), F(10), xn, nTiles);

  // ---- layer 2 (DF=16) ----
  zero_kernel<<<zBlk, 256, 0, stream>>>(agg, N_NODES * 16);
  edge_conv<16><<<eBlk, blk, 0, stream>>>(xn, xe, src, dst, F(11), F(12), F(13), F(14),
                                          xe, agg, eTiles);
  node_conv<16><<<nBlk, blk, 0, stream>>>(xn, agg, F(15), F(16), F(17), F(18), xn, nTiles);

  // ---- layer 3 (DF=16) ----
  zero_kernel<<<zBlk, 256, 0, stream>>>(agg, N_NODES * 16);
  edge_conv<16><<<eBlk, blk, 0, stream>>>(xn, xe, src, dst, F(19), F(20), F(21), F(22),
                                          xe, agg, eTiles);
  node_conv<16><<<nBlk, blk, 0, stream>>>(xn, agg, F(23), F(24), F(25), F(26), xn, nTiles);

  // ---- heads ----
  head_kernel<<<nBlk, blk, 0, stream>>>(xn, F(27), F(28), F(29), F(30), out, nTiles);
  head_kernel<<<eBlk, blk, 0, stream>>>(xe, F(31), F(32), F(33), F(34),
                                        out + (size_t)N_NODES * 3, eTiles);
}
